// TransformerEncoder_52810917872291
// MI455X (gfx1250) — matrix operand order repaired and checked by reading, not compiled, not run on hardware
//
#include <hip/hip_runtime.h>
#include <math.h>

#define B_  4
#define T_  1024
#define D_  1024
#define H_  16
#define HD_ 64
#define L_  4
#define F_  4096
#define KC_ 128
#define G_  16
#define MT_ (B_*T_)    // 4096 rows
#define D3_ (3*D_)     // 3072

typedef unsigned short u16;
typedef u16   u16x8  __attribute__((ext_vector_type(8)));
typedef u16   u16x16 __attribute__((ext_vector_type(16)));
typedef __bf16 bf16_t;
typedef bf16_t v16bf __attribute__((ext_vector_type(16)));
typedef float  v8f   __attribute__((ext_vector_type(8)));

__device__ __forceinline__ u16 f2bf(float f){
  union { float f; unsigned u; } c; c.f = f;
  unsigned u = c.u;
  unsigned r = u + 0x7FFFu + ((u >> 16) & 1u);   // round-to-nearest-even
  return (u16)(r >> 16);
}
__device__ __forceinline__ float gelu_exact(float v){
  return 0.5f * v * (1.0f + erff(v * 0.70710678118654752f));
}
__device__ __forceinline__ v8f vzero8(){
  v8f v;
#pragma unroll
  for (int i=0;i<8;i++) v[i]=0.f;
  return v;
}
__device__ __forceinline__ v16bf combine16(u16x8 lo, u16x8 hi){
  u16x16 r;
#pragma unroll
  for (int i=0;i<8;i++){ r[i]=lo[i]; r[i+8]=hi[i]; }
  return __builtin_bit_cast(v16bf, r);
}
// A-fragment: 16x32 bf16, row-major source; p = &A[row][kbase]; half = lane>>4
// lane<16: elems 0..15 -> K 0..15 ; lane>=16: K 16..31
__device__ __forceinline__ v16bf load_a_frag(const u16* p, int half){
  u16x8 lo = *(const u16x8*)(p + half*16);
  u16x8 hi = *(const u16x8*)(p + half*16 + 8);
  return combine16(lo, hi);
}
// B-fragment: 32x16 bf16 from B^T layout [n][K]; p = &BT[n][kbase + half*16]
__device__ __forceinline__ v16bf load_b_frag(const u16* p){
  u16x8 lo = *(const u16x8*)(p);
  u16x8 hi = *(const u16x8*)(p + 8);
  return combine16(lo, hi);
}
__device__ __forceinline__ v16bf zero_frag(){
  u16x16 r;
#pragma unroll
  for (int i=0;i<16;i++) r[i]=0;
  return __builtin_bit_cast(v16bf, r);
}
__device__ __forceinline__ v8f wmma_bf16(v16bf a, v16bf b, v8f c){
  return __builtin_amdgcn_wmma_f32_16x16x32_bf16(false, a, false, b, (short)0, c, false, false);
}

// ---------------- elementwise / repack kernels ----------------

__global__ __launch_bounds__(256) void cvt_bf_kernel(const float* __restrict__ in, u16* __restrict__ out){
  size_t i = (size_t)blockIdx.x*256 + threadIdx.x;
  out[i] = f2bf(in[i]);
}

// W [K,N] f32 row-major -> WT bf16 [N][K]
__global__ __launch_bounds__(256) void repackT_kernel(const float* __restrict__ W, u16* __restrict__ WT, int K, int N){
  size_t i = (size_t)blockIdx.x*256 + threadIdx.x;
  int k = (int)(i / N);
  int n = (int)(i % N);
  WT[(size_t)n*K + k] = f2bf(W[i]);
}

// conv_w (D, 64, 128) -> WT [dout][tap*64+din], Kdim = 8192
__global__ __launch_bounds__(256) void conv_repack_kernel(const float* __restrict__ W, u16* __restrict__ WT){
  size_t i = (size_t)blockIdx.x*256 + threadIdx.x;   // D*64*128
  int tap  = (int)(i & 127);
  int din  = (int)((i >> 7) & 63);
  int dout = (int)(i >> 13);
  WT[(size_t)dout*8192 + tap*64 + din] = f2bf(W[i]);
}

// xcur = x + gelu(conv_raw + conv_b)
__global__ __launch_bounds__(256) void conv_combine_kernel(const float* __restrict__ x, const float* __restrict__ cr,
                                                           const float* __restrict__ cb, float* __restrict__ xcur){
  size_t i = (size_t)blockIdx.x*256 + threadIdx.x;
  int c = (int)(i & (D_-1));
  xcur[i] = x[i] + gelu_exact(cr[i] + cb[c]);
}

// vT[b,h,d,t] = bf16(qkv[b,t, (2H+h)*64 + d])
__global__ __launch_bounds__(256) void make_vT_kernel(const float* __restrict__ qkv, u16* __restrict__ vT){
  size_t i = (size_t)blockIdx.x*256 + threadIdx.x;   // B*H*HD*T
  int t = (int)(i & (T_-1));
  size_t rest = i >> 10;
  int d = (int)(rest & (HD_-1)); rest >>= 6;
  int h = (int)(rest & (H_-1));
  int b = (int)(rest >> 4);
  float v = qkv[((size_t)(b*T_ + t))*D3_ + (2*H_ + h)*HD_ + d];
  vT[i] = f2bf(v);
}

// gate[b,h,t] = sigmoid(a)*(sigmoid(bb)*c - 1) + 2
__global__ __launch_bounds__(256) void gate_kernel(const float* __restrict__ xin, const float* __restrict__ gw,
                                                   const float* __restrict__ gb, const float* __restrict__ gc,
                                                   float* __restrict__ gate){
  int i = blockIdx.x*256 + threadIdx.x;   // B*H*T
  int t = i & (T_-1);
  int h = (i >> 10) & (H_-1);
  int b = i >> 14;
  const float* gi = xin + ((size_t)(b*T_ + t))*D_ + h*HD_;
  float p[8];
#pragma unroll
  for (int j=0;j<8;j++) p[j] = gb[j];
  for (int d=0; d<HD_; d++){
    float xv = gi[d];
#pragma unroll
    for (int j=0;j<8;j++) p[j] += xv * gw[d*8 + j];
  }
  float a  = p[0]+p[1]+p[2]+p[3];
  float bb = p[4]+p[5]+p[6]+p[7];
  float ga = 1.f/(1.f+expf(-a));
  float gv = 1.f/(1.f+expf(-bb));
  gate[i] = ga*(gv*gc[h] - 1.f) + 2.f;
}

// fused residual + layernorm; one wave per row of D=1024
__global__ __launch_bounds__(256) void ln_kernel(const float* __restrict__ xin, const float* __restrict__ addv,
                                                 const float* __restrict__ sg, const float* __restrict__ bg,
                                                 float* __restrict__ outf, u16* __restrict__ outbf){
  int lane = threadIdx.x & 31, wave = threadIdx.x >> 5;
  int row = blockIdx.x*8 + wave;
  const float* xr = xin + (size_t)row*D_;
  const float* ar = addv ? addv + (size_t)row*D_ : nullptr;
  float v[32];
  float s = 0.f, ss = 0.f;
#pragma unroll
  for (int i=0;i<32;i++){
    int c = lane + i*32;
    float t = xr[c] + (ar ? ar[c] : 0.f);
    v[i] = t; s += t; ss += t*t;
  }
#pragma unroll
  for (int off=16; off>=1; off>>=1){ s += __shfl_xor(s, off, 32); ss += __shfl_xor(ss, off, 32); }
  float mean = s * (1.f/(float)D_);
  float var  = ss * (1.f/(float)D_) - mean*mean;
  float inv  = rsqrtf(var + 1e-5f);
#pragma unroll
  for (int i=0;i<32;i++){
    int c = lane + i*32;
    float y = (v[i]-mean)*inv*sg[c] + bg[c];
    outf[(size_t)row*D_ + c] = y;
    if (outbf) outbf[(size_t)row*D_ + c] = f2bf(y);
  }
}

// ---------------- WMMA GEMM kernels ----------------

// C[M,N] = act(A[M,K]bf16 @ BT[N,K]bf16 + bias)
// 1 wave -> 32x64 tile (2 A-frags x 4 B-frags = 8 WMMA per K-step).
// K-loop unrolled x2 with ping-pong fragment sets: prefetch of one set overlaps
// WMMAs of the other, with no register-rotation moves. Requires K % 64 == 0.
__global__ __launch_bounds__(256) void gemm_bf16_kernel(const u16* __restrict__ A, const u16* __restrict__ BT,
                                                        const float* __restrict__ bias,
                                                        float* __restrict__ Cf, u16* __restrict__ Cbf,
                                                        int M, int N, int K, int act){
  int lane = threadIdx.x & 31, wave = threadIdx.x >> 5;
  int ln = lane & 15, half = lane >> 4;
  int m0 = blockIdx.x * 32;
  int n0 = (blockIdx.y*8 + wave) * 64;
  const u16* arow0 = A + (size_t)(m0 + ln)*K;
  const u16* arow1 = A + (size_t)(m0 + 16 + ln)*K;
  const u16* brow[4];
#pragma unroll
  for (int s=0;s<4;s++) brow[s] = BT + (size_t)(n0 + s*16 + ln)*K + half*16;

  v8f acc[2][4];
#pragma unroll
  for (int b2=0;b2<2;b2++)
#pragma unroll
    for (int s=0;s<4;s++) acc[b2][s] = vzero8();

  // ping-pong fragment sets
  v16bf a0_p, a1_p, b_p[4];   // set P
  v16bf a0_q, a1_q, b_q[4];   // set Q

  // prologue: set P <- K-step @0
  a0_p = load_a_frag(arow0, half);
  a1_p = load_a_frag(arow1, half);
#pragma unroll
  for (int s=0;s<4;s++) b_p[s] = load_b_frag(brow[s]);

#pragma unroll 1
  for (int kk = 0; kk < K - 64; kk += 64){
    // prefetch set Q @ kk+32 while consuming set P @ kk
    a0_q = load_a_frag(arow0 + kk + 32, half);
    a1_q = load_a_frag(arow1 + kk + 32, half);
#pragma unroll
    for (int s=0;s<4;s++) b_q[s] = load_b_frag(brow[s] + kk + 32);
#pragma unroll
    for (int s=0;s<4;s++) acc[0][s] = wmma_bf16(a0_p, b_p[s], acc[0][s]);
#pragma unroll
    for (int s=0;s<4;s++) acc[1][s] = wmma_bf16(a1_p, b_p[s], acc[1][s]);
    // prefetch set P @ kk+64 while consuming set Q @ kk+32
    a0_p = load_a_frag(arow0 + kk + 64, half);
    a1_p = load_a_frag(arow1 + kk + 64, half);
#pragma unroll
    for (int s=0;s<4;s++) b_p[s] = load_b_frag(brow[s] + kk + 64);
#pragma unroll
    for (int s=0;s<4;s++) acc[0][s] = wmma_bf16(a0_q, b_q[s], acc[0][s]);
#pragma unroll
    for (int s=0;s<4;s++) acc[1][s] = wmma_bf16(a1_q, b_q[s], acc[1][s]);
  }
  // epilogue: set P holds @K-64; fetch and consume final step @K-32
  a0_q = load_a_frag(arow0 + K - 32, half);
  a1_q = load_a_frag(arow1 + K - 32, half);
#pragma unroll
  for (int s=0;s<4;s++) b_q[s] = load_b_frag(brow[s] + K - 32);
#pragma unroll
  for (int s=0;s<4;s++) acc[0][s] = wmma_bf16(a0_p, b_p[s], acc[0][s]);
#pragma unroll
  for (int s=0;s<4;s++) acc[1][s] = wmma_bf16(a1_p, b_p[s], acc[1][s]);
#pragma unroll
  for (int s=0;s<4;s++) acc[0][s] = wmma_bf16(a0_q, b_q[s], acc[0][s]);
#pragma unroll
  for (int s=0;s<4;s++) acc[1][s] = wmma_bf16(a1_q, b_q[s], acc[1][s]);

#pragma unroll
  for (int b2=0;b2<2;b2++)
#pragma unroll
    for (int s=0;s<4;s++){
      int col = n0 + s*16 + ln;
      float bv = bias ? bias[col] : 0.f;
#pragma unroll
      for (int r=0;r<8;r++){
        int row = m0 + b2*16 + half*8 + r;
        float v = acc[b2][s][r] + bv;
        if (act == 1) v = gelu_exact(v);
        if (Cf)  Cf[(size_t)row*N + col] = v;
        if (Cbf) Cbf[(size_t)row*N + col] = f2bf(v);
      }
    }
}

// grouped positional conv as GEMM: out[bt, g*64+n] = sum_{tap,din} x[b, t+tap-64, g*64+din]*wT
__global__ __launch_bounds__(128) void conv_gemm_kernel(const u16* __restrict__ xbf, const u16* __restrict__ wT,
                                                        float* __restrict__ out){
  int lane = threadIdx.x & 31, wave = threadIdx.x >> 5;
  int ln = lane & 15, half = lane >> 4;
  int m0 = blockIdx.x*64 + wave*16;
  int g  = blockIdx.y;
  int rIdx = m0 + ln;
  int b = rIdx >> 10, t = rIdx & (T_-1);
  v8f acc[4];
#pragma unroll
  for (int s=0;s<4;s++) acc[s] = vzero8();
#pragma unroll 1
  for (int kk=0; kk<256; ++kk){               // 256 K-chunks of 32 (half-taps)
    int tap  = kk >> 1;
    int doff = (kk & 1) * 32;
    int tin  = t + tap - (KC_/2);
    v16bf a;
    if (tin >= 0 && tin < T_){
      const u16* p = xbf + ((size_t)(b*T_ + tin))*D_ + g*64 + doff;
      a = load_a_frag(p, half);
    } else {
      a = zero_frag();
    }
#pragma unroll
    for (int n=0;n<4;n++){
      const u16* wp = wT + (size_t)(g*64 + n*16 + ln)*8192 + kk*32 + half*16;
      v16bf w = load_b_frag(wp);
      acc[n] = wmma_bf16(a, w, acc[n]);
    }
  }
#pragma unroll
  for (int n=0;n<4;n++)
#pragma unroll
    for (int r=0;r<8;r++)
      out[(size_t)(m0 + half*8 + r)*D_ + g*64 + n*16 + ln] = acc[n][r];
}

// flash attention: 1 wave per (b,h, 16 query rows); online softmax via small LDS tile
__global__ __launch_bounds__(32) void attn_kernel(const u16* __restrict__ qkvbf, const u16* __restrict__ vT,
                                                  const float* __restrict__ pb, const float* __restrict__ gate,
                                                  u16* __restrict__ outbf){
  __shared__ float s_sc[16*32];
  __shared__ u16   s_p [16*32];
  __shared__ float s_m[16], s_l[16], s_scale[16];
  int lane = threadIdx.x;
  int ln = lane & 15, half = lane >> 4;
  int q0 = blockIdx.x * 16;
  int bh = blockIdx.y;
  int b = bh >> 4, h = bh & (H_-1);

  if (lane < 16){ s_m[lane] = -1e30f; s_l[lane] = 0.f; s_scale[lane] = 1.f; }
  __syncthreads();

  // preload Q A-fragments (K = HD = 64 -> 2 frags)
  const u16* qrow = qkvbf + ((size_t)(b*T_ + q0 + ln))*D3_ + h*HD_;
  v16bf qa0 = load_a_frag(qrow,      half);
  v16bf qa1 = load_a_frag(qrow + 32, half);

  float gr[8];
#pragma unroll
  for (int r=0;r<8;r++) gr[r] = gate[(size_t)bh*T_ + q0 + half*8 + r];

  v8f o[4];
#pragma unroll
  for (int n=0;n<4;n++) o[n] = vzero8();

#pragma unroll 1
  for (int kb = 0; kb < T_/32; ++kb){
    // ---- scores for 32 keys ----
#pragma unroll
    for (int st=0; st<2; ++st){
      v8f s = vzero8();
      {
        int key = kb*32 + st*16 + ln;
        const u16* kp0 = qkvbf + ((size_t)(b*T_ + key))*D3_ + (H_ + h)*HD_ + half*16;
        s = wmma_bf16(qa0, load_b_frag(kp0),      s);
        s = wmma_bf16(qa1, load_b_frag(kp0 + 32), s);
      }
#pragma unroll
      for (int r=0;r<8;r++){
        int row = half*8 + r;
        int col = st*16 + ln;
        int key = kb*32 + col;
        float bias = gr[r] * pb[((size_t)h*T_ + q0 + row)*T_ + key];
        s_sc[row*32 + col] = s[r] * 0.125f + bias;   // 1/sqrt(64)
      }
    }
    __syncthreads();
    // ---- online softmax update (lanes 0..15, one row each) ----
    if (lane < 16){
      int row = lane;
      float m_old = s_m[row];
      float mx = m_old;
#pragma unroll
      for (int c=0;c<32;c++) mx = fmaxf(mx, s_sc[row*32+c]);
      float scale = __expf(m_old - mx);
      float lsum = 0.f;
#pragma unroll
      for (int c=0;c<32;c++){
        float p = __expf(s_sc[row*32+c] - mx);
        lsum += p;
        s_p[row*32+c] = f2bf(p);
      }
      s_m[row] = mx;
      s_l[row] = s_l[row]*scale + lsum;
      s_scale[row] = scale;
    }
    __syncthreads();
    // ---- rescale O and accumulate P @ V ----
    float sc[8];
#pragma unroll
    for (int r=0;r<8;r++) sc[r] = s_scale[half*8 + r];
#pragma unroll
    for (int n=0;n<4;n++)
#pragma unroll
      for (int r=0;r<8;r++) o[n][r] *= sc[r];

    v16bf pa = load_a_frag(&s_p[ln*32], half);
#pragma unroll
    for (int n=0;n<4;n++){
      const u16* vp = vT + ((size_t)((b*H_ + h)*HD_ + n*16 + ln))*T_ + kb*32 + half*16;
      o[n] = wmma_bf16(pa, load_b_frag(vp), o[n]);
    }
    __syncthreads();
  }
  // ---- final normalize + store bf16 [b, t, h*64 + d] ----
  float li[8];
#pragma unroll
  for (int r=0;r<8;r++) li[r] = 1.0f / s_l[half*8 + r];
#pragma unroll
  for (int n=0;n<4;n++)
#pragma unroll
    for (int r=0;r<8;r++){
      int row = q0 + half*8 + r;
      int col = h*HD_ + n*16 + ln;
      outbf[(size_t)(b*T_ + row)*D_ + col] = f2bf(o[n][r] * li[r]);
    }
}

// ---------------- host side ----------------

extern "C" void kernel_launch(void* const* d_in, const int* in_sizes, int n_in,
                              void* d_out, int out_size, void* d_ws, size_t ws_size,
                              hipStream_t stream) {
  (void)in_sizes; (void)n_in; (void)out_size; (void)ws_size;
  const float* x      = (const float*)d_in[0];
  const float* pb     = (const float*)d_in[1];
  const float* conv_w = (const float*)d_in[2];
  const float* conv_b = (const float*)d_in[3];
  const float* ln0s   = (const float*)d_in[4];
  const float* ln0b   = (const float*)d_in[5];
  const float* qkv_w  = (const float*)d_in[6];
  const float* qkv_b  = (const float*)d_in[7];
  const float* out_w  = (const float*)d_in[8];
  const float* out_b  = (const float*)d_in[9];
  const float* gru_w  = (const float*)d_in[10];
  const float* gru_b  = (const float*)d_in[11];
  const float* gru_c  = (const float*)d_in[12];
  const float* ln1s   = (const float*)d_in[13];
  const float* ln1b   = (const float*)d_in[14];
  const float* f1w    = (const float*)d_in[15];
  const float* f1b    = (const float*)d_in[16];
  const float* f2w    = (const float*)d_in[17];
  const float* f2b    = (const float*)d_in[18];
  const float* ln2s   = (const float*)d_in[19];
  const float* ln2b   = (const float*)d_in[20];

  char* ws = (char*)d_ws;
  size_t off = 0;
  auto alloc = [&](size_t bytes)->void* {
    void* p = ws + off;
    off += (bytes + 255) & ~(size_t)255;
    return p;
  };
  float* xcur  = (float*)alloc((size_t)MT_*D_*4);
  u16*   xbf   = (u16*)  alloc((size_t)MT_*D_*2);
  float* big   = (float*)alloc((size_t)MT_*F_*4);
  u16*   midbf = (u16*)  alloc((size_t)MT_*F_*2);
  u16*   qkvbf = (u16*)  alloc((size_t)MT_*D3_*2);
  u16*   vT    = (u16*)  alloc((size_t)B_*H_*HD_*T_*2);
  u16*   attnbf= (u16*)  alloc((size_t)MT_*D_*2);
  float* gateb = (float*)alloc((size_t)B_*H_*T_*4);
  u16*   wT    = (u16*)  alloc((size_t)D_*8192*2);   // max: conv (16.8MB) >= F*D (8.4MB)

  // --- positional conv + LN0 ---
  cvt_bf_kernel<<<MT_*D_/256, 256, 0, stream>>>(x, xbf);
  conv_repack_kernel<<<D_*64*128/256, 256, 0, stream>>>(conv_w, wT);
  conv_gemm_kernel<<<dim3(MT_/64, G_), 128, 0, stream>>>(xbf, wT, big);
  conv_combine_kernel<<<MT_*D_/256, 256, 0, stream>>>(x, big, conv_b, xcur);
  ln_kernel<<<MT_/8, 256, 0, stream>>>(xcur, nullptr, ln0s, ln0b, xcur, xbf);

  for (int l = 0; l < L_; ++l){
    // QKV projection (bf16 copy written directly for Q/K fragments)
    repackT_kernel<<<D_*D3_/256, 256, 0, stream>>>(qkv_w + (size_t)l*D_*D3_, wT, D_, D3_);
    gemm_bf16_kernel<<<dim3(MT_/32, D3_/512), 256, 0, stream>>>(xbf, wT, qkv_b + (size_t)l*D3_,
                                                                big, qkvbf, MT_, D3_, D_, 0);
    make_vT_kernel<<<B_*H_*HD_*T_/256, 256, 0, stream>>>(big, vT);
    gate_kernel<<<B_*H_*T_/256, 256, 0, stream>>>(xcur, gru_w + (size_t)l*HD_*8,
                                                  gru_b + (size_t)l*8, gru_c + (size_t)l*H_, gateb);
    // flash attention, writes bf16 attn output [b,t,D]
    attn_kernel<<<dim3(T_/16, B_*H_), 32, 0, stream>>>(qkvbf, vT, pb, gateb, attnbf);
    // output projection
    repackT_kernel<<<D_*D_/256, 256, 0, stream>>>(out_w + (size_t)l*D_*D_, wT, D_, D_);
    gemm_bf16_kernel<<<dim3(MT_/32, D_/512), 256, 0, stream>>>(attnbf, wT, out_b + (size_t)l*D_,
                                                               big, nullptr, MT_, D_, D_, 0);
    // residual + LN1
    ln_kernel<<<MT_/8, 256, 0, stream>>>(xcur, big, ln1s + (size_t)l*D_, ln1b + (size_t)l*D_, xcur, xbf);
    // FFN
    repackT_kernel<<<D_*F_/256, 256, 0, stream>>>(f1w + (size_t)l*D_*F_, wT, D_, F_);
    gemm_bf16_kernel<<<dim3(MT_/32, F_/512), 256, 0, stream>>>(xbf, wT, f1b + (size_t)l*F_,
                                                               nullptr, midbf, MT_, F_, D_, 1);
    repackT_kernel<<<F_*D_/256, 256, 0, stream>>>(f2w + (size_t)l*F_*D_, wT, F_, D_);
    gemm_bf16_kernel<<<dim3(MT_/32, D_/512), 256, 0, stream>>>(midbf, wT, f2b + (size_t)l*D_,
                                                               big, nullptr, MT_, D_, F_, 0);
    // residual + LN2 (last layer writes straight to d_out)
    float* dst = (l == L_-1) ? (float*)d_out : xcur;
    ln_kernel<<<MT_/8, 256, 0, stream>>>(xcur, big, ln2s + (size_t)l*D_, ln2b + (size_t)l*D_, dst, xbf);
  }
}
